// GraphEncoderLayer_50646254354666
// MI455X (gfx1250) — compile-verified, matrix operand order checked by reading, not run on hardware
//
#include <hip/hip_runtime.h>
#include <math.h>

// ---------------------------------------------------------------------------
// Problem constants (match reference)
// ---------------------------------------------------------------------------
constexpr int Bc = 4, Nn = 4096, Kk = 32, Cc = 128, CEc = 128, Hh = 4, HDd = 32, Ee = 128;
constexpr float LN_EPS = 1e-5f;
constexpr float INV_SQRT_HD = 0.17677669529663687f; // 1/sqrt(32)

// ---------------------------------------------------------------------------
// bf16 helpers (raw-bit representation in LDS; RNE float->bf16)
// ---------------------------------------------------------------------------
typedef __attribute__((ext_vector_type(16))) __bf16 v16bf;
typedef __attribute__((ext_vector_type(8)))  float  v8f;

__device__ __forceinline__ unsigned short f2bfbits(float f) {
  union { float f; unsigned u; } a; a.f = f;
  unsigned u = a.u;
  unsigned r = u + 0x7FFFu + ((u >> 16) & 1u);   // round-to-nearest-even
  return (unsigned short)(r >> 16);
}
__device__ __forceinline__ float bf2f(unsigned short h) {
  union { unsigned u; float f; } a; a.u = ((unsigned)h) << 16;
  return a.f;
}
__device__ __forceinline__ float gelu_exact(float h) {
  return 0.5f * h * (1.0f + erff(h * 0.70710678118654752f));
}

// ---------------------------------------------------------------------------
// WMMA fragment loader.
// ISA 7.12.2 (16-bit operands, wave32): for A (16xK tile, row striped in lane)
// and B stored transposed in LDS as [n][k], each lane holds 16 bf16 = 8 dwords.
// lane&15 selects row (A: M) / column (B: N); lane>>4 selects the K-half:
//   VGPR v<4 : k = 2v   + 8*half (+kOff)
//   VGPR v>=4: k = 16+2(v-4) + 8*half (+kOff)
// "row" passed in already includes the (lane&15) component.
// ---------------------------------------------------------------------------
__device__ __forceinline__ v16bf load_frag(const unsigned short* base, int row,
                                           int strideHW, int kOff, int lane) {
  const int half = lane >> 4;
  union { v16bf v; unsigned u[8]; } r;
  const unsigned short* p = base + row * strideHW + kOff;
#pragma unroll
  for (int v = 0; v < 8; ++v) {
    int k = (v < 4) ? (2 * v + 8 * half) : (16 + 2 * (v - 4) + 8 * half);
    r.u[v] = *(const unsigned*)(p + k);
  }
  return r.v;
}

// ---------------------------------------------------------------------------
// LDS budgets (dynamic; gfx1250 allows up to 320KB per workgroup)
// ---------------------------------------------------------------------------
constexpr int SB_HW   = 256 * 264;   // WEKV^T bf16, padded stride (bank-friendly)
constexpr int SWQA_HW = 128 * 128;   // WQa bf16
constexpr int SWOA_HW = 128 * 128;   // WOa bf16
constexpr int SA_HW   = 32 * 264;    // per-node A tile bf16
constexpr int ATTN_HW = SB_HW + SWQA_HW + SWOA_HW + SA_HW;
constexpr int ATTN_F  = 32 * 256 /*skv*/ + 256 /*bEKV*/ + 128 /*q*/ + 36 /*mask*/
                        + 132 /*log*/ + 132 /*attn*/ + 128 /*out*/;
constexpr size_t SMEM_ATTN = (size_t)ATTN_HW * 2 + (size_t)ATTN_F * 4; // ~253.6 KB

constexpr int SW_HW   = 512 * 136;   // max(W1^T [512][136], W2^T [128][520]) in hw
constexpr int SXLN_HW = 32 * 136;    // LN1 activations bf16
constexpr int SH_HW   = 32 * 520;    // hidden activations bf16
constexpr int MLP_HW  = SW_HW + SXLN_HW + SH_HW;
constexpr int MLP_F   = 32 * 128 /*res*/ + 64 /*mean,inv*/ + 512 /*b1*/ + 128 /*b2*/;
constexpr size_t SMEM_MLP = (size_t)MLP_HW * 2 + (size_t)MLP_F * 4;    // ~200.4 KB

// ===========================================================================
// Pool attention, step 1: qp = pool_vec @ WQp + bQp        grid(B) x 32
// ===========================================================================
__global__ void __launch_bounds__(32) pool_q_kernel(
    const float* __restrict__ pool_vec, const float* __restrict__ WQp,
    const float* __restrict__ bQp, float* __restrict__ ws_qp) {
  int b = blockIdx.x, t = threadIdx.x;
  float s = bQp[t];
  for (int c = 0; c < Cc; ++c) s += pool_vec[b * Cc + c] * WQp[c * HDd + t];
  ws_qp[b * HDd + t] = s;
}

// ===========================================================================
// Pool attention, step 2: per-row k,v and masked logits.    grid(B*64) x 64
// Row j=0 is pool_vec itself; rows 1..N are x.
// ===========================================================================
__global__ void __launch_bounds__(64) pool_logits_kernel(
    const float* __restrict__ x, const float* __restrict__ pool_vec,
    const float* __restrict__ x_mask, const float* __restrict__ WKVp,
    const float* __restrict__ bKVp, const float* __restrict__ ws_qp,
    float* __restrict__ ws_logits, float* __restrict__ ws_vp) {
  __shared__ float kp[HDd];
  const int b = blockIdx.x >> 6;
  const int chunk = blockIdx.x & 63;
  const int tid = threadIdx.x;
  const int d = tid & 31, half = tid >> 5;
  const int j0 = chunk * 65;
  for (int jj = 0; jj < 65; ++jj) {
    const int j = j0 + jj;
    if (j > Nn) break;  // uniform across block
    const float* row = (j == 0) ? (pool_vec + b * Cc)
                                : (x + ((long)b * Nn + (j - 1)) * Cc);
    float s = bKVp[half * HDd + d];
    for (int c = 0; c < Cc; ++c) s += row[c] * WKVp[c * 2 * HDd + half * HDd + d];
    if (half == 0) kp[d] = s;
    else ws_vp[((long)b * (Nn + 1) + j) * HDd + d] = s;
    __syncthreads();
    if (tid == 0) {
      float lg = 0.f;
      for (int d2 = 0; d2 < HDd; ++d2) lg += kp[d2] * ws_qp[b * HDd + d2];
      float mk = (j == 0) ? 1.f : x_mask[(long)b * Nn + (j - 1)];
      lg = (mk == 0.f) ? -3.0e38f : lg;
      ws_logits[b * (Nn + 1) + j] = lg * INV_SQRT_HD;
    }
    __syncthreads();
  }
}

// ===========================================================================
// Pool attention, step 3: softmax + pooled V + output proj + pvp + kv_pool.
// grid(B) x 256
// ===========================================================================
__global__ void __launch_bounds__(256) pool_finish_kernel(
    const float* __restrict__ pool_vec, const float* __restrict__ WOp,
    const float* __restrict__ bOp, const float* __restrict__ Wpvp,
    const float* __restrict__ bpvp, const float* __restrict__ WEKV,
    const float* __restrict__ bEKV, const float* __restrict__ ws_logits,
    const float* __restrict__ ws_vp, float* __restrict__ ws_pool,
    float* __restrict__ ws_kvpool) {
  __shared__ float red[256];
  __shared__ float pooled[HDd];
  __shared__ float spool[Cc];
  __shared__ float spvp[CEc + Cc];
  const int b = blockIdx.x, tid = threadIdx.x;
  const float* lg = ws_logits + b * (Nn + 1);
  // max
  float mx = -3.0e38f;
  for (int j = tid; j <= Nn; j += 256) mx = fmaxf(mx, lg[j]);
  red[tid] = mx; __syncthreads();
  for (int s = 128; s > 0; s >>= 1) { if (tid < s) red[tid] = fmaxf(red[tid], red[tid + s]); __syncthreads(); }
  mx = red[0]; __syncthreads();
  // sum of exp + per-thread partial pooled-V
  float se = 0.f, acc[HDd];
#pragma unroll
  for (int d2 = 0; d2 < HDd; ++d2) acc[d2] = 0.f;
  for (int j = tid; j <= Nn; j += 256) {
    float w = __expf(lg[j] - mx);
    se += w;
    const float* vp = ws_vp + ((long)b * (Nn + 1) + j) * HDd;
#pragma unroll
    for (int d2 = 0; d2 < HDd; ++d2) acc[d2] += w * vp[d2];
  }
  red[tid] = se; __syncthreads();
  for (int s = 128; s > 0; s >>= 1) { if (tid < s) red[tid] += red[tid + s]; __syncthreads(); }
  const float inv = 1.f / red[0]; __syncthreads();
  for (int d2 = 0; d2 < HDd; ++d2) {
    red[tid] = acc[d2]; __syncthreads();
    for (int s = 128; s > 0; s >>= 1) { if (tid < s) red[tid] += red[tid + s]; __syncthreads(); }
    if (tid == 0) pooled[d2] = red[0] * inv;
    __syncthreads();
  }
  // pool = pool_vec + pooled @ WOp + bOp
  if (tid < Cc) {
    float s = pool_vec[b * Cc + tid] + bOp[tid];
    for (int d2 = 0; d2 < HDd; ++d2) s += pooled[d2] * WOp[d2 * Cc + tid];
    spool[tid] = s;
    ws_pool[b * Cc + tid] = s;
  }
  __syncthreads();
  // pvp = pool @ Wpvp + bpvp
  {
    float s = bpvp[tid];
    for (int c = 0; c < Cc; ++c) s += spool[c] * Wpvp[c * (CEc + Cc) + tid];
    spvp[tid] = s;
  }
  __syncthreads();
  // kv_pool = pvp @ WEKV + bEKV
  {
    float s = bEKV[tid];
    for (int c = 0; c < CEc + Cc; ++c) s += spvp[c] * WEKV[c * 2 * Ee + tid];
    ws_kvpool[b * 2 * Ee + tid] = s;
  }
}

// ===========================================================================
// Main fused per-node attention kernel (WMMA bf16).  grid(B*128) x 256,
// each workgroup owns 32 nodes and stages WEKV^T/WQa/WOa in LDS once.
// ===========================================================================
__global__ void __launch_bounds__(256) concat_attn_kernel(
    const float* __restrict__ x, const int* __restrict__ E_idx,
    const float* __restrict__ E_features, const float* __restrict__ e_mask,
    const float* __restrict__ WQa, const float* __restrict__ bQa,
    const float* __restrict__ WOa, const float* __restrict__ bOa,
    const float* __restrict__ WEKV, const float* __restrict__ bEKV,
    const float* __restrict__ ws_kvpool, float* __restrict__ out) {
  extern __shared__ unsigned char smem[];
  unsigned short* sB   = (unsigned short*)smem;        // WEKV^T bf16 [n=256][k pad 264]
  unsigned short* sWQa = sB + SB_HW;                   // [c=128][e=128]
  unsigned short* sWOa = sWQa + SWQA_HW;               // [e=128][c=128]
  unsigned short* sA   = sWOa + SWOA_HW;               // [r=32][k pad 264]
  float* skv   = (float*)(sA + SA_HW);                 // [32][256]  (K cols 0..127, V 128..255)
  float* sbEKV = skv + 32 * 256;                       // [256]
  float* sq    = sbEKV + 256;                          // [128]
  float* smask = sq + 128;                             // [33] (+pad)
  float* slog  = smask + 36;                           // [4][33]
  float* sattn = slog + 132;                           // [4][33]
  float* sout  = sattn + 132;                          // [128]

  const int tid  = threadIdx.x;
  const int b    = blockIdx.x >> 7;    // /128
  const int wg   = blockIdx.x & 127;
  const int wave = tid >> 5, lane = tid & 31;
  const int nl   = lane & 15, half = lane >> 4;

  // ---- one-time staging per workgroup ----
  for (int i = tid; i < 256 * 256; i += 256) {
    int n = i >> 8, k = i & 255;
    sB[n * 264 + k] = f2bfbits(WEKV[k * 256 + n]);     // transpose to [n][k]
  }
  for (int i = tid; i < 128 * 128; i += 256) {
    sWQa[i] = f2bfbits(WQa[i]);
    sWOa[i] = f2bfbits(WOa[i]);
  }
  sbEKV[tid] = bEKV[tid];
  __syncthreads();

  for (int ni = 0; ni < 32; ++ni) {
    const int node = wg * 32 + ni;
    const long nb = (long)b * Nn + node;

    // ---- gather A tile [32 rows][gathered-x 128 | E_features 128] -> bf16 LDS
    for (int i = tid; i < 32 * 256; i += 256) {
      int r = i >> 8, c = i & 255;
      float v;
      if (c < Cc) v = x[((long)b * Nn + E_idx[nb * Kk + r]) * Cc + c];
      else        v = E_features[(nb * Kk + r) * CEc + (c - Cc)];
      sA[r * 264 + c] = f2bfbits(v);
    }
    if (tid < Kk) smask[tid + 1] = e_mask[nb * Kk + tid];
    if (tid == 0) smask[0] = 1.0f;
    if (tid < Ee) {          // q = x_row @ WQa + bQa
      float s = bQa[tid];
      for (int c = 0; c < Cc; ++c) s += x[nb * Cc + c] * bf2f(sWQa[c * Ee + tid]);
      sq[tid] = s;
    }
    if (ni + 1 < 32 && tid == 0)
      __builtin_prefetch(&E_features[(nb + 1) * Kk * CEc], 0, 1);
    __syncthreads();

    // ---- kv[32][256] = A @ WEKV  via v_wmma_f32_16x16x32_bf16 ----
    for (int t = 0; t < 2; ++t) {
      const int nt = wave * 2 + t;          // 8 waves x 2 tiles = 16 N-tiles
      v8f acc0 = {}; v8f acc1 = {};
#pragma unroll
      for (int kt = 0; kt < 8; ++kt) {
        v16bf a0 = load_frag(sA, nl,       264, kt * 32, lane);   // rows 0..15
        v16bf a1 = load_frag(sA, 16 + nl,  264, kt * 32, lane);   // rows 16..31
        v16bf bb = load_frag(sB, nt * 16 + nl, 264, kt * 32, lane);
        acc0 = __builtin_amdgcn_wmma_f32_16x16x32_bf16(false, a0, false, bb,
                                                       (short)0, acc0, false, false);
        acc1 = __builtin_amdgcn_wmma_f32_16x16x32_bf16(false, a1, false, bb,
                                                       (short)0, acc1, false, false);
      }
      const int ncol = nt * 16 + nl;
      const float bias = sbEKV[ncol];
#pragma unroll
      for (int r = 0; r < 8; ++r) {         // D layout: lane=N, vgpr r => M=r+8*half
        skv[(half * 8 + r) * 256 + ncol]        = acc0[r] + bias;
        skv[(16 + half * 8 + r) * 256 + ncol]   = acc1[r] + bias;
      }
    }
    __syncthreads();

    // ---- logits over K+1 = 33 (row 0 = pool) ----
    if (tid < Hh * 33) {
      int h = tid / 33, r = tid - h * 33;
      const float* kr = (r == 0) ? (ws_kvpool + b * 2 * Ee + h * HDd)
                                 : (skv + (r - 1) * 256 + h * HDd);
      float s = 0.f;
      for (int d = 0; d < HDd; ++d) s += sq[h * HDd + d] * kr[d];
      s = (smask[r] == 0.f) ? -3.0e38f : s;
      slog[h * 33 + r] = s * INV_SQRT_HD;
    }
    __syncthreads();
    if (tid < Hh) {                         // tiny softmax per head
      float mx = -3.0e38f;
      for (int r = 0; r < 33; ++r) mx = fmaxf(mx, slog[tid * 33 + r]);
      float se = 0.f;
      for (int r = 0; r < 33; ++r) {
        float e = __expf(slog[tid * 33 + r] - mx);
        sattn[tid * 33 + r] = e; se += e;
      }
      float inv = 1.f / se;
      for (int r = 0; r < 33; ++r) sattn[tid * 33 + r] *= inv;
    }
    __syncthreads();
    if (tid < Ee) {                         // out = attn @ V
      int h = tid >> 5, d = tid & 31;
      float s = sattn[h * 33] * ws_kvpool[b * 2 * Ee + Ee + h * HDd + d];
      for (int r = 1; r < 33; ++r)
        s += sattn[h * 33 + r] * skv[(r - 1) * 256 + Ee + h * HDd + d];
      sout[tid] = s;
    }
    __syncthreads();
    if (tid < Cc) {                         // x + out @ WOa + bOa
      float y = x[nb * Cc + tid] + bOa[tid];
      for (int e = 0; e < Ee; ++e) y += sout[e] * bf2f(sWOa[e * Cc + tid]);
      out[nb * Cc + tid] = y;
    }
    __syncthreads();
  }
}

// ===========================================================================
// Node MLP: in-place on out rows.  grid(16384/32) x 256.
// h = LN1(x); h = h + gelu(h@W1+b1)@W2 + b2; out = LN2(h)   — WMMA bf16 GEMMs.
// ===========================================================================
__global__ void __launch_bounds__(256) mlp_kernel(
    float* __restrict__ out, const float* __restrict__ W1,
    const float* __restrict__ b1, const float* __restrict__ W2,
    const float* __restrict__ b2, const float* __restrict__ g1,
    const float* __restrict__ be1, const float* __restrict__ g2,
    const float* __restrict__ be2) {
  extern __shared__ unsigned char smem[];
  unsigned short* sW   = (unsigned short*)smem;        // W1^T [512][136] or W2^T [128][520]
  unsigned short* sXln = sW + SW_HW;                   // [32][136]
  unsigned short* sH   = sXln + SXLN_HW;               // [32][520]
  float* sXres = (float*)(sH + SH_HW);                 // [32][128]  (LN1 out, then result)
  float* smv   = sXres + 32 * 128;                     // [32][2] mean / inv-std
  float* sb1   = smv + 64;                             // [512]
  float* sb2   = sb1 + 512;                            // [128]

  const int tid = threadIdx.x;
  const int row0 = blockIdx.x * 32;
  const int wave = tid >> 5, lane = tid & 31;
  const int nl = lane & 15, half = lane >> 4;

  for (int i = tid; i < 512; i += 256) sb1[i] = b1[i];
  if (tid < 128) sb2[tid] = b2[tid];
  for (int i = tid; i < 32 * 128; i += 256) sXres[i] = out[(long)row0 * 128 + i];
  __syncthreads();
  // ---- LN1 ----
  if (tid < 32) {
    float m = 0.f;
    for (int c = 0; c < 128; ++c) m += sXres[tid * 128 + c];
    m *= (1.f / 128.f);
    float v = 0.f;
    for (int c = 0; c < 128; ++c) { float d = sXres[tid * 128 + c] - m; v += d * d; }
    smv[tid * 2] = m;
    smv[tid * 2 + 1] = rsqrtf(v * (1.f / 128.f) + LN_EPS);
  }
  __syncthreads();
  for (int i = tid; i < 32 * 128; i += 256) {
    int r = i >> 7, c = i & 127;
    float ln = (sXres[i] - smv[r * 2]) * smv[r * 2 + 1] * g1[c] + be1[c];
    sXres[i] = ln;                       // residual source = LN1 output
    sXln[r * 136 + c] = f2bfbits(ln);
  }
  // ---- stage W1^T bf16 ----
  for (int i = tid; i < 512 * 128; i += 256) {
    int n = i >> 7, k = i & 127;
    sW[n * 136 + k] = f2bfbits(W1[k * 512 + n]);
  }
  __syncthreads();
  // ---- GEMM1 [32,128]@[128,512] + gelu -> sH ----
  for (int t = 0; t < 4; ++t) {
    const int nt = wave * 4 + t;             // 32 N-tiles
    v8f acc0 = {}; v8f acc1 = {};
#pragma unroll
    for (int kt = 0; kt < 4; ++kt) {
      v16bf a0 = load_frag(sXln, nl,      136, kt * 32, lane);
      v16bf a1 = load_frag(sXln, 16 + nl, 136, kt * 32, lane);
      v16bf bb = load_frag(sW, nt * 16 + nl, 136, kt * 32, lane);
      acc0 = __builtin_amdgcn_wmma_f32_16x16x32_bf16(false, a0, false, bb,
                                                     (short)0, acc0, false, false);
      acc1 = __builtin_amdgcn_wmma_f32_16x16x32_bf16(false, a1, false, bb,
                                                     (short)0, acc1, false, false);
    }
    const int ncol = nt * 16 + nl;
    const float bias = sb1[ncol];
#pragma unroll
    for (int r = 0; r < 8; ++r) {
      sH[(half * 8 + r) * 520 + ncol]      = f2bfbits(gelu_exact(acc0[r] + bias));
      sH[(16 + half * 8 + r) * 520 + ncol] = f2bfbits(gelu_exact(acc1[r] + bias));
    }
  }
  __syncthreads();
  // ---- restage W2^T bf16 (same buffer) ----
  for (int i = tid; i < 128 * 512; i += 256) {
    int n = i >> 9, k = i & 511;
    sW[n * 520 + k] = f2bfbits(W2[k * 128 + n]);
  }
  __syncthreads();
  // ---- GEMM2 [32,512]@[512,128] + residual -> sXres ----
  {
    const int nt = wave;                     // 8 N-tiles
    v8f acc0 = {}; v8f acc1 = {};
#pragma unroll
    for (int kt = 0; kt < 16; ++kt) {
      v16bf a0 = load_frag(sH, nl,      520, kt * 32, lane);
      v16bf a1 = load_frag(sH, 16 + nl, 520, kt * 32, lane);
      v16bf bb = load_frag(sW, nt * 16 + nl, 520, kt * 32, lane);
      acc0 = __builtin_amdgcn_wmma_f32_16x16x32_bf16(false, a0, false, bb,
                                                     (short)0, acc0, false, false);
      acc1 = __builtin_amdgcn_wmma_f32_16x16x32_bf16(false, a1, false, bb,
                                                     (short)0, acc1, false, false);
    }
    const int ncol = nt * 16 + nl;
    const float bias = sb2[ncol];
#pragma unroll
    for (int r = 0; r < 8; ++r) {
      sXres[(half * 8 + r) * 128 + ncol]      += acc0[r] + bias;
      sXres[(16 + half * 8 + r) * 128 + ncol] += acc1[r] + bias;
    }
  }
  __syncthreads();
  // ---- LN2 + writeback ----
  if (tid < 32) {
    float m = 0.f;
    for (int c = 0; c < 128; ++c) m += sXres[tid * 128 + c];
    m *= (1.f / 128.f);
    float v = 0.f;
    for (int c = 0; c < 128; ++c) { float d = sXres[tid * 128 + c] - m; v += d * d; }
    smv[tid * 2] = m;
    smv[tid * 2 + 1] = rsqrtf(v * (1.f / 128.f) + LN_EPS);
  }
  __syncthreads();
  for (int i = tid; i < 32 * 128; i += 256) {
    int r = i >> 7, c = i & 127;
    out[(long)row0 * 128 + i] = (sXres[i] - smv[r * 2]) * smv[r * 2 + 1] * g2[c] + be2[c];
  }
}

// ===========================================================================
// Pool MLP (4 rows, VALU).  grid(B) x 128 -> writes tail of d_out.
// ===========================================================================
__global__ void __launch_bounds__(128) pool_mlp_kernel(
    const float* __restrict__ ws_pool, const float* __restrict__ W1,
    const float* __restrict__ b1, const float* __restrict__ W2,
    const float* __restrict__ b2, const float* __restrict__ g1,
    const float* __restrict__ be1, const float* __restrict__ g2,
    const float* __restrict__ be2, float* __restrict__ out_tail) {
  __shared__ float red[128], sln[128], sh[512];
  const int b = blockIdx.x, tid = threadIdx.x;
  float p = ws_pool[b * Cc + tid];
  red[tid] = p; __syncthreads();
  for (int s = 64; s > 0; s >>= 1) { if (tid < s) red[tid] += red[tid + s]; __syncthreads(); }
  float m = red[0] * (1.f / 128.f); __syncthreads();
  float d0 = p - m;
  red[tid] = d0 * d0; __syncthreads();
  for (int s = 64; s > 0; s >>= 1) { if (tid < s) red[tid] += red[tid + s]; __syncthreads(); }
  float inv = rsqrtf(red[0] * (1.f / 128.f) + LN_EPS); __syncthreads();
  float ln = d0 * inv * g1[tid] + be1[tid];
  sln[tid] = ln; __syncthreads();
  for (int j = tid; j < 512; j += 128) {
    float s = b1[j];
    for (int c = 0; c < 128; ++c) s += sln[c] * W1[c * 512 + j];
    sh[j] = gelu_exact(s);
  }
  __syncthreads();
  float y = ln + b2[tid];
  for (int j = 0; j < 512; ++j) y += sh[j] * W2[j * 128 + tid];
  red[tid] = y; __syncthreads();
  for (int s = 64; s > 0; s >>= 1) { if (tid < s) red[tid] += red[tid + s]; __syncthreads(); }
  float m2 = red[0] * (1.f / 128.f); __syncthreads();
  float d2 = y - m2;
  red[tid] = d2 * d2; __syncthreads();
  for (int s = 64; s > 0; s >>= 1) { if (tid < s) red[tid] += red[tid + s]; __syncthreads(); }
  float inv2 = rsqrtf(red[0] * (1.f / 128.f) + LN_EPS);
  out_tail[b * Cc + tid] = d2 * inv2 * g2[tid] + be2[tid];
}

// ===========================================================================
// Launch
// ===========================================================================
extern "C" void kernel_launch(void* const* d_in, const int* in_sizes, int n_in,
                              void* d_out, int out_size, void* d_ws, size_t ws_size,
                              hipStream_t stream) {
  (void)in_sizes; (void)n_in; (void)out_size; (void)ws_size;
  const float* x          = (const float*)d_in[0];
  const int*   E_idx      = (const int*)d_in[1];
  const float* E_features = (const float*)d_in[2];
  const float* e_mask     = (const float*)d_in[3];
  const float* x_mask     = (const float*)d_in[4];
  const float* pool_vec   = (const float*)d_in[5];
  const float* WQa = (const float*)d_in[6];  const float* bQa = (const float*)d_in[7];
  const float* WOa = (const float*)d_in[8];  const float* bOa = (const float*)d_in[9];
  const float* Wpvp = (const float*)d_in[10]; const float* bpvp = (const float*)d_in[11];
  const float* WEKV = (const float*)d_in[12]; const float* bEKV = (const float*)d_in[13];
  const float* WQp = (const float*)d_in[14]; const float* bQp = (const float*)d_in[15];
  const float* WOp = (const float*)d_in[16]; const float* bOp = (const float*)d_in[17];
  const float* WKVp = (const float*)d_in[18]; const float* bKVp = (const float*)d_in[19];
  const float* W1 = (const float*)d_in[20]; const float* b1 = (const float*)d_in[21];
  const float* W2 = (const float*)d_in[22]; const float* b2 = (const float*)d_in[23];
  const float* g1 = (const float*)d_in[24]; const float* be1 = (const float*)d_in[25];
  const float* g2 = (const float*)d_in[26]; const float* be2 = (const float*)d_in[27];
  float* out = (float*)d_out;

  // workspace layout (floats); requires ~2.2 MB
  float* ws         = (float*)d_ws;
  float* ws_pool    = ws;                                   // [B][C]
  float* ws_kvpool  = ws + Bc * Cc;                         // [B][2E]
  float* ws_qp      = ws_kvpool + Bc * 2 * Ee;              // [B][HD]
  float* ws_logits  = ws_qp + Bc * HDd;                     // [B][N+1]
  float* ws_vp      = ws_logits + Bc * (Nn + 1);            // [B][N+1][HD]

  // allow >64KB dynamic LDS (idempotent, deterministic)
  hipFuncSetAttribute(reinterpret_cast<const void*>(concat_attn_kernel),
                      hipFuncAttributeMaxDynamicSharedMemorySize, (int)SMEM_ATTN);
  hipFuncSetAttribute(reinterpret_cast<const void*>(mlp_kernel),
                      hipFuncAttributeMaxDynamicSharedMemorySize, (int)SMEM_MLP);

  pool_q_kernel<<<dim3(Bc), dim3(32), 0, stream>>>(pool_vec, WQp, bQp, ws_qp);
  pool_logits_kernel<<<dim3(Bc * 64), dim3(64), 0, stream>>>(
      x, pool_vec, x_mask, WKVp, bKVp, ws_qp, ws_logits, ws_vp);
  pool_finish_kernel<<<dim3(Bc), dim3(256), 0, stream>>>(
      pool_vec, WOp, bOp, Wpvp, bpvp, WEKV, bEKV, ws_logits, ws_vp,
      ws_pool, ws_kvpool);
  concat_attn_kernel<<<dim3(Bc * 128), dim3(256), SMEM_ATTN, stream>>>(
      x, E_idx, E_features, e_mask, WQa, bQa, WOa, bOa, WEKV, bEKV,
      ws_kvpool, out);
  mlp_kernel<<<dim3((Bc * Nn) / 32), dim3(256), SMEM_MLP, stream>>>(
      out, W1, b1, W2, b2, g1, be1, g2, be2);
  pool_mlp_kernel<<<dim3(Bc), dim3(128), 0, stream>>>(
      ws_pool, W1, b1, W2, b2, g1, be1, g2, be2, out + (long)Bc * Nn * Cc);
}
